// MultiHeadAttention_43482248904868
// MI455X (gfx1250) — compile-verified, matrix operand order checked by reading, not run on hardware
//
#include <hip/hip_runtime.h>
#include <hip/hip_bf16.h>
#include <stddef.h>

// ---------------------------------------------------------------------------
// MI455X (gfx1250) MHA kernel. Wave32, WMMA bf16 16x16x32 w/ f32 accumulate.
// Roofline: ~70 GFLOP of GEMM (trivial for WMMA) vs ~1.07 GB of forced HBM
// traffic for the returned attention tensor (write P once, read once for PV)
// => memory-bound at ~46us @ 23.3 TB/s. All GEMMs routed through
// v_wmma_f32_16x16x32_bf16; intermediates held in bf16 scratch (~67 MB < L2).
// This rev: f32->bf16 via v_cvt_pk_bf16_f32 (1 VALU / 2 elems instead of the
// 3-VALU/elem bit trick) + global_prefetch_b8 on streaming GEMM A-operands.
// ---------------------------------------------------------------------------

#define B_  16
#define L_  1024
#define D_  512
#define H_  8
#define DK_ 64
#define DV_ 64

typedef __bf16 bf16_t;
typedef __attribute__((ext_vector_type(16))) __bf16       v16bf;
typedef __attribute__((ext_vector_type(2)))  __bf16       v2bf;
typedef __attribute__((ext_vector_type(8)))  float        v8f;
typedef __attribute__((ext_vector_type(4)))  unsigned int v4u;

#if defined(__has_builtin)
#if __has_builtin(__builtin_amdgcn_cvt_pk_bf16_f32)
#define HAVE_PK_BF16 1
#endif
#endif

__device__ __forceinline__ v2bf pk2(float a, float b) {
#ifdef HAVE_PK_BF16
  return __builtin_amdgcn_cvt_pk_bf16_f32(a, b);
#else
  v2bf r;
  r[0] = (__bf16)a;
  r[1] = (__bf16)b;
  return r;
#endif
}

__device__ __forceinline__ bf16_t f2bf(float f) {
#ifdef HAVE_PK_BF16
  v2bf t = __builtin_amdgcn_cvt_pk_bf16_f32(f, f);
  return t[0];
#else
  return (bf16_t)f;
#endif
}

__device__ __forceinline__ v8f wmma_bf16(v16bf a, v16bf b, v8f c) {
  // (neg_a, A, neg_b, B, c_mod, C, reuse_a, reuse_b)
  return __builtin_amdgcn_wmma_f32_16x16x32_bf16(false, a, false, b, (short)0, c,
                                                 false, false);
}

// 16x32 bf16 fragment (A operand, or B operand as rows of W = columns of W^T).
// ISA 7.12.2: lane<16 -> row=lane, elems 0..7 = K0..7, 8..15 = K16..23;
//             lane>=16 -> row=lane-16, elems = K8..15, K24..31.
__device__ __forceinline__ v16bf load_frag_bf(const bf16_t* __restrict__ base, int ld) {
  int lane = threadIdx.x & 31;
  int half = lane >> 4, r = lane & 15;
  const bf16_t* p0 = base + r * ld + half * 8;
  const bf16_t* p1 = p0 + 16;
  union { v4u u[2]; v16bf v; } t;
  t.u[0] = *reinterpret_cast<const v4u*>(p0);
  t.u[1] = *reinterpret_cast<const v4u*>(p1);
  return t.v;
}

// Same fragment but sourced from f32 memory, converted in-register via
// packed bf16 converts.
__device__ __forceinline__ v16bf load_frag_f32(const float* __restrict__ base, int ld) {
  int lane = threadIdx.x & 31;
  int half = lane >> 4, r = lane & 15;
  const float* p0 = base + r * ld + half * 8;
  const float* p1 = p0 + 16;
  v16bf o;
#pragma unroll
  for (int i = 0; i < 4; ++i) {
    v2bf a = pk2(p0[2 * i], p0[2 * i + 1]);
    v2bf b = pk2(p1[2 * i], p1[2 * i + 1]);
    o[2 * i]     = a[0];
    o[2 * i + 1] = a[1];
    o[8 + 2 * i]     = b[0];
    o[8 + 2 * i + 1] = b[1];
  }
  return o;
}

// ---------------------------------------------------------------------------
// Bias precompute: td = x/mean(x); td += softplus(tp)*max(td); bias = tm/log(e+td)
// ---------------------------------------------------------------------------
__global__ __launch_bounds__(256) void bias_kernel(
    const float* __restrict__ time_diff, const float* __restrict__ time_plus,
    const float* __restrict__ time_mul, float* __restrict__ bias) {
  __shared__ float sred[256];
  const int t = threadIdx.x;
  const int N = B_ * L_;
  float s = 0.f;
  for (int i = t; i < N; i += 256) s += time_diff[i];
  sred[t] = s;
  __syncthreads();
  for (int k = 128; k > 0; k >>= 1) {
    if (t < k) sred[t] += sred[t + k];
    __syncthreads();
  }
  const float mean = sred[0] / (float)N;
  __syncthreads();
  float mx = -1e30f;
  for (int i = t; i < N; i += 256) mx = fmaxf(mx, time_diff[i]);
  sred[t] = mx;
  __syncthreads();
  for (int k = 128; k > 0; k >>= 1) {
    if (t < k) sred[t] = fmaxf(sred[t], sred[t + k]);
    __syncthreads();
  }
  const float maxtd = sred[0] / mean;
  const float sp = log1pf(expf(time_plus[0]));   // softplus
  const float tm = time_mul[0];
  const float EC = 2.7182818284590452f;
  for (int i = t; i < N; i += 256) {
    float td = time_diff[i] / mean + sp * maxtd;
    bias[i] = tm / logf(EC + td);
  }
}

// ---------------------------------------------------------------------------
// Projection: C[BL,512] = X[BL,512] @ W[512,512]^T, scattered to per-head bf16.
// mode 0: out[b][h][l][dk]   mode 1 (V): out[b][h][dv][l] (transposed)
// Block = 256 thr = 8 waves; block covers 16 rows x 512 cols (wave: 16x64).
// ---------------------------------------------------------------------------
__global__ __launch_bounds__(256) void proj_kernel(
    const float* __restrict__ X, const float* __restrict__ W,
    bf16_t* __restrict__ out, int mode) {
  const int m0 = blockIdx.x * 16;
  const int wave = threadIdx.x >> 5;
  const int lane = threadIdx.x & 31, half = lane >> 4, r = lane & 15;
  const int n0w = wave * 64;
  v8f acc[4] = {};
  for (int k0 = 0; k0 < D_; k0 += 32) {
    if (k0 + 64 < D_)
      __builtin_prefetch(X + m0 * D_ + k0 + 64 + (lane & 15) * D_, 0, 0);
    v16bf a = load_frag_f32(X + m0 * D_ + k0, D_);
#pragma unroll
    for (int f = 0; f < 4; ++f) {
      v16bf w = load_frag_f32(W + (n0w + f * 16) * D_ + k0, D_);
      acc[f] = wmma_bf16(a, w, acc[f]);
    }
  }
#pragma unroll
  for (int f = 0; f < 4; ++f) {
    const int col = n0w + f * 16 + r;
    const int h = col >> 6, dd = col & 63;
#pragma unroll
    for (int i = 0; i < 8; ++i) {
      const int row = m0 + i + 8 * half;
      const int bb = row >> 10;       // / L_
      const int ll = row & 1023;      // % L_
      bf16_t v = f2bf(acc[f][i]);
      if (mode == 0)
        out[((bb * H_ + h) * L_ + ll) * 64 + dd] = v;
      else
        out[((bb * H_ + h) * 64 + dd) * L_ + ll] = v;
    }
  }
}

// ---------------------------------------------------------------------------
// Attention scores + softmax. Block = one (b,h) x 16 query rows x all 1024 keys.
// Wave w handles keys [w*128, w*128+128) -> 8 C-frags. P (f32) -> d_out attn.
// ---------------------------------------------------------------------------
__global__ __launch_bounds__(256) void attn_kernel(
    const bf16_t* __restrict__ qh, const bf16_t* __restrict__ kh,
    const float* __restrict__ bias, const unsigned char* __restrict__ mask,
    float* __restrict__ attn_out) {
  const int QT = L_ / 16;
  const int qt = blockIdx.x % QT;
  const int hb = blockIdx.x / QT;
  const int h = hb % H_, b = hb / H_;
  const int q0 = qt * 16;
  const int wave = threadIdx.x >> 5;
  const int lane = threadIdx.x & 31, half = lane >> 4, r = lane & 15;

  const bf16_t* Q = qh + (size_t)(b * H_ + h) * L_ * DK_;
  const bf16_t* K = kh + (size_t)(b * H_ + h) * L_ * DK_;

  v16bf a0 = load_frag_bf(Q + q0 * DK_ + 0, DK_);
  v16bf a1 = load_frag_bf(Q + q0 * DK_ + 32, DK_);

  v8f acc[8] = {};
  const int nbase = wave * 128;
#pragma unroll
  for (int f = 0; f < 8; ++f) {
    const int n0 = nbase + f * 16;
    v16bf w0 = load_frag_bf(K + n0 * DK_ + 0, DK_);
    v16bf w1 = load_frag_bf(K + n0 * DK_ + 32, DK_);
    acc[f] = wmma_bf16(a0, w0, acc[f]);
    acc[f] = wmma_bf16(a1, w1, acc[f]);
  }

  // scale (1/sqrt(64)) + time bias + mask
  const float scale = 0.125f;
#pragma unroll
  for (int f = 0; f < 8; ++f) {
    const int col = nbase + f * 16 + r;
    const float bv = bias[b * L_ + col];
#pragma unroll
    for (int i = 0; i < 8; ++i) {
      const int qrow = q0 + i + 8 * half;
      float s = acc[f][i] * scale + bv;
      if (mask[(size_t)(b * L_ + qrow) * L_ + col]) s = -1e10f;
      acc[f][i] = s;
    }
  }

  // ---- row softmax across 1024 keys (in-lane -> 16-lane half -> waves) ----
  __shared__ float red[16][8];
  float rowmax[8], inv[8];

  float m8[8];
#pragma unroll
  for (int i = 0; i < 8; ++i) {
    float m = acc[0][i];
#pragma unroll
    for (int f = 1; f < 8; ++f) m = fmaxf(m, acc[f][i]);
#pragma unroll
    for (int off = 1; off < 16; off <<= 1) m = fmaxf(m, __shfl_xor(m, off));
    m8[i] = m;
  }
  if (r == 0) {
#pragma unroll
    for (int i = 0; i < 8; ++i) red[i + 8 * half][wave] = m8[i];
  }
  __syncthreads();
#pragma unroll
  for (int i = 0; i < 8; ++i) {
    float m = red[i + 8 * half][0];
#pragma unroll
    for (int w = 1; w < 8; ++w) m = fmaxf(m, red[i + 8 * half][w]);
    rowmax[i] = m;
  }
  __syncthreads();

  float s8[8];
#pragma unroll
  for (int i = 0; i < 8; ++i) {
    float s = 0.f;
#pragma unroll
    for (int f = 0; f < 8; ++f) {
      float e = __expf(acc[f][i] - rowmax[i]);
      acc[f][i] = e;
      s += e;
    }
#pragma unroll
    for (int off = 1; off < 16; off <<= 1) s += __shfl_xor(s, off);
    s8[i] = s;
  }
  if (r == 0) {
#pragma unroll
    for (int i = 0; i < 8; ++i) red[i + 8 * half][wave] = s8[i];
  }
  __syncthreads();
#pragma unroll
  for (int i = 0; i < 8; ++i) {
    float s = 0.f;
#pragma unroll
    for (int w = 0; w < 8; ++w) s += red[i + 8 * half][w];
    inv[i] = 1.0f / s;
  }

  // store P (f32) in torch layout [h*B+b][q][k]
  float* dst = attn_out + (size_t)(h * B_ + b) * L_ * L_;
#pragma unroll
  for (int f = 0; f < 8; ++f) {
    const int col = nbase + f * 16 + r;
#pragma unroll
    for (int i = 0; i < 8; ++i) {
      const int qrow = q0 + i + 8 * half;
      dst[(size_t)qrow * L_ + col] = acc[f][i] * inv[i];
    }
  }
}

// ---------------------------------------------------------------------------
// O = P @ V. P f32 (from d_out attn region, bf16-converted in-register),
// V^T bf16 [b][h][dv][L]. Wave: 16 q-rows x 64 dv, K-loop over 1024.
// Output bf16 in [B*L, H*DV] layout for the FC GEMM.
// ---------------------------------------------------------------------------
__global__ __launch_bounds__(256) void pv_kernel(
    const float* __restrict__ attn, const bf16_t* __restrict__ vhT,
    bf16_t* __restrict__ obf) {
  const int MT = L_ / 128;
  const int mt = blockIdx.x % MT;
  const int hb = blockIdx.x / MT;
  const int h = hb % H_, b = hb / H_;
  const int wave = threadIdx.x >> 5;
  const int lane = threadIdx.x & 31, half = lane >> 4, r = lane & 15;
  const int q0 = mt * 128 + wave * 16;

  const float* P = attn + (size_t)(h * B_ + b) * L_ * L_;
  const bf16_t* V = vhT + (size_t)(b * H_ + h) * DV_ * L_;

  v8f acc[4] = {};
  for (int k0 = 0; k0 < L_; k0 += 32) {
    if (k0 + 64 < L_)
      __builtin_prefetch(P + (size_t)(q0 + (lane & 15)) * L_ + k0 + 64, 0, 0);
    v16bf a = load_frag_f32(P + (size_t)q0 * L_ + k0, L_);
#pragma unroll
    for (int f = 0; f < 4; ++f) {
      v16bf w = load_frag_bf(V + (f * 16) * L_ + k0, L_);
      acc[f] = wmma_bf16(a, w, acc[f]);
    }
  }
#pragma unroll
  for (int f = 0; f < 4; ++f) {
    const int dv = f * 16 + r;
#pragma unroll
    for (int i = 0; i < 8; ++i) {
      const int qrow = q0 + i + 8 * half;
      obf[(size_t)(b * L_ + qrow) * (H_ * DV_) + h * DV_ + dv] = f2bf(acc[f][i]);
    }
  }
}

// ---------------------------------------------------------------------------
// out = O @ fc_w^T + fc_b  (f32 result into d_out[0 : B*L*D], LN applied after)
// ---------------------------------------------------------------------------
__global__ __launch_bounds__(256) void fc_kernel(
    const bf16_t* __restrict__ obf, const float* __restrict__ fcw,
    const float* __restrict__ fcb, float* __restrict__ out) {
  const int m0 = blockIdx.x * 16;
  const int wave = threadIdx.x >> 5;
  const int lane = threadIdx.x & 31, half = lane >> 4, r = lane & 15;
  const int n0w = wave * 64;
  v8f acc[4] = {};
  for (int k0 = 0; k0 < D_; k0 += 32) {
    if (k0 + 64 < D_)
      __builtin_prefetch(obf + (size_t)(m0 + (lane & 15)) * D_ + k0 + 64, 0, 0);
    v16bf a = load_frag_bf(obf + (size_t)m0 * D_ + k0, D_);
#pragma unroll
    for (int f = 0; f < 4; ++f) {
      v16bf w = load_frag_f32(fcw + (n0w + f * 16) * D_ + k0, D_);
      acc[f] = wmma_bf16(a, w, acc[f]);
    }
  }
#pragma unroll
  for (int f = 0; f < 4; ++f) {
    const int col = n0w + f * 16 + r;
    const float bv = fcb[col];
#pragma unroll
    for (int i = 0; i < 8; ++i) {
      out[(size_t)(m0 + i + 8 * half) * D_ + col] = acc[f][i] + bv;
    }
  }
}

// ---------------------------------------------------------------------------
// In-place LayerNorm over last dim (512). One block per row.
// ---------------------------------------------------------------------------
__global__ __launch_bounds__(256) void ln_kernel(
    float* __restrict__ out, const float* __restrict__ g,
    const float* __restrict__ bta) {
  float* p = out + (size_t)blockIdx.x * D_;
  const int t = threadIdx.x;
  const float x0 = p[t], x1 = p[t + 256];
  __shared__ float ssum[256], ssq[256];
  ssum[t] = x0 + x1;
  ssq[t] = x0 * x0 + x1 * x1;
  __syncthreads();
  for (int s = 128; s > 0; s >>= 1) {
    if (t < s) {
      ssum[t] += ssum[t + s];
      ssq[t] += ssq[t + s];
    }
    __syncthreads();
  }
  const float mu = ssum[0] * (1.0f / D_);
  const float var = ssq[0] * (1.0f / D_) - mu * mu;
  const float invs = rsqrtf(var + 1e-5f);
  p[t] = (x0 - mu) * invs * g[t] + bta[t];
  p[t + 256] = (x1 - mu) * invs * g[t + 256] + bta[t + 256];
}

// ---------------------------------------------------------------------------
extern "C" void kernel_launch(void* const* d_in, const int* in_sizes, int n_in,
                              void* d_out, int out_size, void* d_ws, size_t ws_size,
                              hipStream_t stream) {
  (void)in_sizes; (void)n_in; (void)out_size; (void)ws_size;
  const float* q  = (const float*)d_in[0];
  const float* k  = (const float*)d_in[1];
  const float* v  = (const float*)d_in[2];
  const float* td = (const float*)d_in[3];
  const unsigned char* mask = (const unsigned char*)d_in[4];   // jnp bool = 1 byte
  const float* Wq = (const float*)d_in[5];
  const float* Wk = (const float*)d_in[6];
  const float* Wv = (const float*)d_in[7];
  const float* tp = (const float*)d_in[8];
  const float* tm = (const float*)d_in[9];
  const float* fcw = (const float*)d_in[10];
  const float* fcb = (const float*)d_in[11];
  const float* lng = (const float*)d_in[12];
  const float* lnb = (const float*)d_in[13];

  float* out_f = (float*)d_out;                       // [B*L, D]
  float* attn  = out_f + (size_t)B_ * L_ * D_;        // [H*B, L, L]

  // workspace layout
  char* w = (char*)d_ws;
  float* bias = (float*)w;                                   // B*L f32
  bf16_t* qh  = (bf16_t*)(w + 65536);                        // [b][h][l][dk]
  bf16_t* kh  = qh + (size_t)B_ * H_ * L_ * DK_;             // [b][h][l][dk]
  bf16_t* vhT = kh + (size_t)B_ * H_ * L_ * DK_;             // [b][h][dv][l]
  bf16_t* obf = vhT + (size_t)B_ * H_ * L_ * DV_;            // [B*L, H*DV]

  bias_kernel<<<1, 256, 0, stream>>>(td, tp, tm, bias);

  const int proj_blocks = (B_ * L_) / 16;   // 1024
  proj_kernel<<<proj_blocks, 256, 0, stream>>>(q, Wq, qh, 0);
  proj_kernel<<<proj_blocks, 256, 0, stream>>>(k, Wk, kh, 0);
  proj_kernel<<<proj_blocks, 256, 0, stream>>>(v, Wv, vhT, 1);

  attn_kernel<<<B_ * H_ * (L_ / 16), 256, 0, stream>>>(qh, kh, bias, mask, attn);

  pv_kernel<<<B_ * H_ * (L_ / 128), 256, 0, stream>>>(attn, vhT, obf);

  fc_kernel<<<(B_ * L_) / 16, 256, 0, stream>>>(obf, fcw, fcb, out_f);

  ln_kernel<<<B_ * L_, 256, 0, stream>>>(out_f, lng, lnb);
}
